// GNNYield_876173328577
// MI455X (gfx1250) — matrix operand order repaired and checked by reading, not compiled, not run on hardware
//
#include <hip/hip_runtime.h>
#include <hip/hip_bf16.h>

#define N_NODES 100000
#define N_EDGES 3200000
#define N_GRAPHS 4096
#define IN_DIM 128
#define HID 64
#define EPS 1e-5f
#define LDSK (IN_DIM + 16)   // padded k-stride for transposed W in LDS (mult of 16 halves)

typedef __attribute__((ext_vector_type(16))) _Float16 v16h;
typedef __attribute__((ext_vector_type(8)))  _Float16 v8h;
typedef __attribute__((ext_vector_type(8)))  float    v8f;

// ---------------- small utility kernels ----------------
__global__ void k_zero(float* __restrict__ p, int n) {
    int i = blockIdx.x * blockDim.x + threadIdx.x;
    if (i < n) p[i] = 0.f;
}
__global__ void k_fill1(float* __restrict__ p, int n) {
    int i = blockIdx.x * blockDim.x + threadIdx.x;
    if (i < n) p[i] = 1.f;   // self-loop contributes 1 to every node's degree
}
__global__ void k_deg_acc(const int* __restrict__ col, float* __restrict__ deg, int e) {
    int i = blockIdx.x * blockDim.x + threadIdx.x;
    if (i < e) atomicAdd(&deg[col[i]], 1.f);
}
__global__ void k_rsqrt_inplace(float* __restrict__ p, int n) {
    int i = blockIdx.x * blockDim.x + threadIdx.x;
    if (i < n) { float d = p[i]; p[i] = (d > 0.f) ? rsqrtf(d) : 0.f; }
}
__global__ void k_f2h(const float* __restrict__ src, _Float16* __restrict__ dst, long n) {
    long i = blockIdx.x * (long)blockDim.x + threadIdx.x;
    long stride = (long)gridDim.x * blockDim.x;
    for (; i < n; i += stride) dst[i] = (_Float16)src[i];
}

// ---------------- WMMA GEMM: C[M x 64] = A[M x K] @ W[K x 64] + bias ----------------
// blockDim = 256 (8 waves). Block covers 32 rows: wave w -> m-tile (blockIdx*2 + w/4),
// n-tile (w&3)*16. K in {64,128}, M multiple of 32 -> no predication (EXEC all ones
// around v_wmma, as the ISA requires).
__global__ void __launch_bounds__(256) k_gemm(const _Float16* __restrict__ A,
                                              const _Float16* __restrict__ W,
                                              const float* __restrict__ bias,
                                              float* __restrict__ C, int M, int K) {
    __shared__ _Float16 ldsWT[HID * LDSK];   // W transposed: ldsWT[n*LDSK + k]
    for (int i = threadIdx.x; i < K * HID; i += 256) {
        int k = i >> 6, n = i & (HID - 1);
        ldsWT[n * LDSK + k] = W[i];          // coalesced global read, scattered LDS write
    }
    __syncthreads();

    const int wave   = threadIdx.x >> 5;
    const int lane   = threadIdx.x & 31;
    const int half16 = lane >> 4;            // which 16-lane half of the wave
    const int l16    = lane & 15;
    const int m0     = (blockIdx.x * 2 + (wave >> 2)) * 16;
    const int n0     = (wave & 3) * 16;
    const int mrow   = m0 + l16;

    v8f acc = {};
    for (int k0 = 0; k0 < K; k0 += 32) {
        // A fragment (16x32 f16, ISA layout): elem i -> k = k0 + 8*half16 + 16*(i>>3) + (i&7)
        const v8h* ap = (const v8h*)(A + (size_t)mrow * K + k0 + half16 * 8);
        v8h alo = ap[0];       // k .. k+7
        v8h ahi = ap[2];       // k+16 .. k+23
        v16h a;
#pragma unroll
        for (int i = 0; i < 8; ++i) { a[i] = alo[i]; a[8 + i] = ahi[i]; }
        // B fragment (32x16 f16): col = n0+l16, elem i -> k = k0 + 8*half16 + 16*(i>>3) + (i&7)
        const v8h* bp = (const v8h*)(ldsWT + (size_t)(n0 + l16) * LDSK + k0 + half16 * 8);
        v8h blo = bp[0];
        v8h bhi = bp[2];
        v16h b;
#pragma unroll
        for (int i = 0; i < 8; ++i) { b[i] = blo[i]; b[8 + i] = bhi[i]; }
        acc = __builtin_amdgcn_wmma_f32_16x16x32_f16(false, a, false, b,
                                                     (short)0, acc, false, false);
    }
    // C/D layout: VGPR r, lanes0-15 -> row m0+r, lanes16-31 -> row m0+8+r, col = n0+l16
    const float bv   = bias[n0 + l16];
    const int   crow = m0 + half16 * 8;
#pragma unroll
    for (int r = 0; r < 8; ++r)
        C[(size_t)(crow + r) * HID + n0 + l16] = acc[r] + bv;
}

// ---------------- message passing ----------------
// out[i][:] = hw[i][:] * dinv[i]^2   (self-loop term; also initializes out)
__global__ void k_selfinit(const float* __restrict__ hw, const float* __restrict__ dinv,
                           float* __restrict__ out) {
    int idx = blockIdx.x * 256 + threadIdx.x;
    if (idx >= N_NODES * HID) return;
    float d = dinv[idx >> 6];
    out[idx] = hw[idx] * d * d;
}
// 32 lanes per edge, 2 floats per lane: out[col] += hw[row] * dinv[row]*dinv[col]
__global__ void __launch_bounds__(256) k_edges(const int* __restrict__ row,
                                               const int* __restrict__ col,
                                               const float* __restrict__ dinv,
                                               const float* __restrict__ hw,
                                               float* __restrict__ out) {
    int t = blockIdx.x * 256 + threadIdx.x;
    int e = t >> 5;
    if (e >= N_EDGES) return;
    int lane = t & 31;
    int r = row[e], c = col[e];
    float nrm = dinv[r] * dinv[c];
    float2 v = ((const float2*)(hw + (size_t)r * HID))[lane];
    float* dst = out + (size_t)c * HID + lane * 2;
    atomicAdd(dst,     v.x * nrm);
    atomicAdd(dst + 1, v.y * nrm);
}

// ---------------- batchnorm (stats must be pre-zeroed: stats[0:64]=sum, [64:128]=sumsq) ----
__global__ void __launch_bounds__(256) k_bn_stats(const float* __restrict__ h,
                                                  float* __restrict__ stats, int n) {
    __shared__ float ssum[HID], ssq[HID];
    int c = threadIdx.x & (HID - 1);
    int g = threadIdx.x >> 6;        // 4 row-groups per block
    if (threadIdx.x < HID) { ssum[threadIdx.x] = 0.f; ssq[threadIdx.x] = 0.f; }
    __syncthreads();
    float s = 0.f, q = 0.f;
    for (int i = blockIdx.x * 4 + g; i < n; i += gridDim.x * 4) {
        float v = h[(size_t)i * HID + c];
        s += v; q += v * v;
    }
    atomicAdd(&ssum[c], s);
    atomicAdd(&ssq[c], q);
    __syncthreads();
    if (threadIdx.x < HID) {
        atomicAdd(&stats[threadIdx.x],       ssum[threadIdx.x]);
        atomicAdd(&stats[HID + threadIdx.x], ssq[threadIdx.x]);
    }
}
__global__ void k_bn_relu(float* __restrict__ h, const float* __restrict__ stats,
                          const float* __restrict__ gam, const float* __restrict__ bet, int n) {
    long idx = blockIdx.x * 256L + threadIdx.x;
    if (idx >= (long)n * HID) return;
    int c = (int)(idx & (HID - 1));
    float inv_n = 1.0f / (float)n;
    float mu  = stats[c] * inv_n;
    float var = stats[HID + c] * inv_n - mu * mu;
    float v = (h[idx] - mu) * rsqrtf(var + EPS) * gam[c] + bet[c];
    h[idx] = v > 0.f ? v : 0.f;
}

// ---------------- global mean pool ----------------
__global__ void k_pool(const float* __restrict__ h, const int* __restrict__ batch,
                       float* __restrict__ pooled, float* __restrict__ cnt) {
    int t = blockIdx.x * 256 + threadIdx.x;
    int nidx = t >> 5;
    if (nidx >= N_NODES) return;
    int lane = t & 31;
    int g = batch[nidx];
    float2 v = ((const float2*)(h + (size_t)nidx * HID))[lane];
    float* dst = pooled + (size_t)g * HID + lane * 2;
    atomicAdd(dst,     v.x);
    atomicAdd(dst + 1, v.y);
    if (lane == 0) atomicAdd(&cnt[g], 1.0f);
}
__global__ void k_pool_div(float* __restrict__ pooled, const float* __restrict__ cnt) {
    int idx = blockIdx.x * 256 + threadIdx.x;
    if (idx >= N_GRAPHS * HID) return;
    pooled[idx] /= fmaxf(cnt[idx >> 6], 1.0f);
}

// ---------------- final head: out[g] = dot(z[g,:], W2) + b2 ----------------
__global__ void __launch_bounds__(256) k_head(const float* __restrict__ z,
                                              const float* __restrict__ W2,
                                              const float* __restrict__ b2,
                                              float* __restrict__ out) {
    int g = blockIdx.x * 8 + (threadIdx.x >> 5);
    int lane = threadIdx.x & 31;
    const float* zr = z + (size_t)g * HID;
    float acc = zr[lane] * W2[lane] + zr[lane + 32] * W2[lane + 32];
#pragma unroll
    for (int off = 16; off > 0; off >>= 1) acc += __shfl_down(acc, off, 32);
    if (lane == 0) out[g] = acc + b2[0];
}

// ---------------- driver ----------------
extern "C" void kernel_launch(void* const* d_in, const int* in_sizes, int n_in,
                              void* d_out, int out_size, void* d_ws, size_t ws_size,
                              hipStream_t stream) {
    const float* x     = (const float*)d_in[0];
    const int*   ei    = (const int*)d_in[1];
    const int*   row   = ei;
    const int*   col   = ei + N_EDGES;
    const int*   batch = (const int*)d_in[2];
    const float* Wc0 = (const float*)d_in[3],  *bc0 = (const float*)d_in[4];
    const float* g0  = (const float*)d_in[5],  *be0 = (const float*)d_in[6];
    const float* Wc1 = (const float*)d_in[7],  *bc1 = (const float*)d_in[8];
    const float* g1  = (const float*)d_in[9],  *be1 = (const float*)d_in[10];
    const float* Wc2 = (const float*)d_in[11], *bc2 = (const float*)d_in[12];
    const float* g2  = (const float*)d_in[13], *be2 = (const float*)d_in[14];
    const float* W1  = (const float*)d_in[15], *b1  = (const float*)d_in[16];
    const float* gf  = (const float*)d_in[17], *bf  = (const float*)d_in[18];
    const float* W2  = (const float*)d_in[19], *b2  = (const float*)d_in[20];
    float* out = (float*)d_out;

    char* ws = (char*)d_ws;
    size_t off = 0;
    auto alloc = [&](size_t bytes) -> char* {
        char* p = ws + off; off += (bytes + 255) & ~(size_t)255; return p;
    };
    float*    dinv   = (float*)alloc((size_t)N_NODES * 4);
    _Float16* ah     = (_Float16*)alloc((size_t)N_NODES * IN_DIM * 2);
    float*    bufA   = (float*)alloc((size_t)N_NODES * HID * 4);   // GEMM out (hW)
    float*    bufB   = (float*)alloc((size_t)N_NODES * HID * 4);   // conv out / layer act
    _Float16* wh     = (_Float16*)alloc((size_t)(IN_DIM * HID + 3 * HID * HID) * 2);
    float*    stats  = (float*)alloc((size_t)4 * 2 * HID * 4);     // 4 BN invocations
    float*    pooled = (float*)alloc((size_t)N_GRAPHS * HID * 4);
    float*    cnt    = (float*)alloc((size_t)N_GRAPHS * 4);
    _Float16* ph     = (_Float16*)alloc((size_t)N_GRAPHS * HID * 2);
    float*    zbuf   = (float*)alloc((size_t)N_GRAPHS * HID * 4);

    _Float16* Wc0h = wh;
    _Float16* Wc1h = Wc0h + IN_DIM * HID;
    _Float16* Wc2h = Wc1h + HID * HID;
    _Float16* W1h  = Wc2h + HID * HID;

    // zero accumulators (ws is poisoned; must be re-zeroed every call)
    k_zero<<<2,   256, 0, stream>>>(stats, 4 * 2 * HID);
    k_zero<<<(N_GRAPHS * HID + 255) / 256, 256, 0, stream>>>(pooled, N_GRAPHS * HID);
    k_zero<<<(N_GRAPHS + 255) / 256,       256, 0, stream>>>(cnt, N_GRAPHS);

    // deg (with self-loops) -> dinv = rsqrt(deg)
    k_fill1<<<(N_NODES + 255) / 256, 256, 0, stream>>>(dinv, N_NODES);
    k_deg_acc<<<(N_EDGES + 255) / 256, 256, 0, stream>>>(col, dinv, N_EDGES);
    k_rsqrt_inplace<<<(N_NODES + 255) / 256, 256, 0, stream>>>(dinv, N_NODES);

    // weights -> fp16
    k_f2h<<<32, 256, 0, stream>>>(Wc0, Wc0h, IN_DIM * HID);
    k_f2h<<<16, 256, 0, stream>>>(Wc1, Wc1h, HID * HID);
    k_f2h<<<16, 256, 0, stream>>>(Wc2, Wc2h, HID * HID);
    k_f2h<<<16, 256, 0, stream>>>(W1,  W1h,  HID * HID);

    const float* Ws[3]   = {nullptr, nullptr, nullptr};  // fp16 already
    _Float16*    Whs[3]  = {Wc0h, Wc1h, Wc2h};
    const float* bs[3]   = {bc0, bc1, bc2};
    const float* gs[3]   = {g0, g1, g2};
    const float* bes[3]  = {be0, be1, be2};
    (void)Ws;

    for (int layer = 0; layer < 3; ++layer) {
        int K = (layer == 0) ? IN_DIM : HID;
        const float* src = (layer == 0) ? x : bufB;
        k_f2h<<<4096, 256, 0, stream>>>(src, ah, (long)N_NODES * K);
        k_gemm<<<N_NODES / 32, 256, 0, stream>>>(ah, Whs[layer], bs[layer], bufA, N_NODES, K);
        k_selfinit<<<(N_NODES * HID) / 256, 256, 0, stream>>>(bufA, dinv, bufB);
        k_edges<<<(N_EDGES * 32) / 256, 256, 0, stream>>>(row, col, dinv, bufA, bufB);
        k_bn_stats<<<512, 256, 0, stream>>>(bufB, stats + layer * 128, N_NODES);
        k_bn_relu<<<(N_NODES * HID) / 256, 256, 0, stream>>>(bufB, stats + layer * 128,
                                                             gs[layer], bes[layer], N_NODES);
    }

    // global mean pool
    k_pool<<<(N_NODES * 32 + 255) / 256, 256, 0, stream>>>(bufB, batch, pooled, cnt);
    k_pool_div<<<(N_GRAPHS * HID) / 256, 256, 0, stream>>>(pooled, cnt);

    // MLP head: pooled @ W1 + b1 -> BN -> relu -> @ W2 + b2
    k_f2h<<<64, 256, 0, stream>>>(pooled, ph, (long)N_GRAPHS * HID);
    k_gemm<<<N_GRAPHS / 32, 256, 0, stream>>>(ph, W1h, b1, zbuf, N_GRAPHS, HID);
    k_bn_stats<<<64, 256, 0, stream>>>(zbuf, stats + 384, N_GRAPHS);
    k_bn_relu<<<(N_GRAPHS * HID) / 256, 256, 0, stream>>>(zbuf, stats + 384, gf, bf, N_GRAPHS);
    k_head<<<N_GRAPHS / 8, 256, 0, stream>>>(zbuf, W2, b2, out);
}
